// EnhancedProgressiveRefinementModule_70205535421036
// MI455X (gfx1250) — matrix operand order repaired and checked by reading, not compiled, not run on hardware
//
#include <hip/hip_runtime.h>
#include <hip/hip_bf16.h>
#include <cstdint>
#include <cstddef>

// ---------------- problem constants (from setup_inputs) ----------------
#define BATCH 4
#define CIN   768
#define CH    256      // HIDDEN
#define CRQ   64       // HIDDEN/4
#define HH    32
#define WWW   32
#define HW    1024
#define NQ    4        // quarters
#define NSTG  3        // STAGES
#define BNM   0.9999950000374997f   // 1/sqrt(1+1e-5)

typedef __attribute__((ext_vector_type(16))) __bf16        v16bf;
typedef __attribute__((ext_vector_type(8)))  float         v8f;
typedef __attribute__((ext_vector_type(4)))  unsigned int  u32x4;
typedef __attribute__((ext_vector_type(8)))  int           i32x8;
typedef __attribute__((ext_vector_type(4)))  int           i32x4;

#if defined(__has_builtin)
# if __has_builtin(__builtin_amdgcn_tensor_load_to_lds) && __has_builtin(__builtin_amdgcn_s_wait_tensorcnt)
#  define HAVE_TDM 1
# endif
#endif
#ifndef HAVE_TDM
# define HAVE_TDM 0
#endif

// ---------------- small helpers ----------------
__device__ inline unsigned short f2bfu(float f) {           // f32 -> bf16 bits (RNE)
  unsigned u = __float_as_uint(f);
  unsigned r = u + 0x7fffu + ((u >> 16) & 1u);
  return (unsigned short)(r >> 16);
}
__device__ inline v8f vzero() {
  v8f v;
#pragma unroll
  for (int i = 0; i < 8; i++) v[i] = 0.f;
  return v;
}
__device__ inline v8f wmma_bf16(v16bf a, v16bf b, v8f c) {
  return __builtin_amdgcn_wmma_f32_16x16x32_bf16(false, a, false, b, (short)0, c, false, false);
}
// A-matrix 16x32 bf16 fragment: row = lane%16; K = (lane/16)*16 + e (contiguous run, matches B)
__device__ inline v16bf load_afrag(const unsigned short* __restrict__ row, int lgrp) {
  v16bf a;
#pragma unroll
  for (int e = 0; e < 16; e++) {
    int k = lgrp * 16 + e;
    a[e] = __builtin_bit_cast(__bf16, row[k]);
  }
  return a;
}
// B-matrix 32x16 bf16 fragment: col = lane%16; K = (lane/16)*16 + e (contiguous run)
__device__ inline v16bf load_bfrag(const unsigned short* __restrict__ colbase, int lgrp) {
  v16bf b;
  const unsigned short* p = colbase + lgrp * 16;
#pragma unroll
  for (int e = 0; e < 16; e++) b[e] = __builtin_bit_cast(__bf16, p[e]);
  return b;
}

// ---------------- mask statistics (DS atomics) ----------------
__global__ void k_stats(const float* __restrict__ mask, int* __restrict__ stats) {
  __shared__ int s_cnt, s_sy, s_sx, s_mny, s_mxy, s_mnx, s_mxx;
  const int b = blockIdx.x;
  if (threadIdx.x == 0) { s_cnt = 0; s_sy = 0; s_sx = 0; s_mny = 1 << 30; s_mxy = -1; s_mnx = 1 << 30; s_mxx = -1; }
  __syncthreads();
  for (int p = threadIdx.x; p < HW; p += blockDim.x) {
    if (mask[b * HW + p] > 0.5f) {
      int y = p >> 5, x = p & 31;
      atomicAdd(&s_cnt, 1); atomicAdd(&s_sy, y); atomicAdd(&s_sx, x);
      atomicMin(&s_mny, y); atomicMax(&s_mxy, y);
      atomicMin(&s_mnx, x); atomicMax(&s_mxx, x);
    }
  }
  __syncthreads();
  if (threadIdx.x == 0) {
    int* o = stats + b * 8;
    o[0] = s_cnt; o[1] = s_sy; o[2] = s_sx; o[3] = s_mny; o[4] = s_mxy; o[5] = s_mnx; o[6] = s_mxx;
  }
}

// plan layout (ints): [0..2]=active, [3..5]=scale s, [6..101]=starts[3][B][4][2]
__global__ void k_plan(const int* __restrict__ stats, int* __restrict__ plan) {
  if (threadIdx.x != 0 || blockIdx.x != 0) return;
  float cy[BATCH], cx[BATCH];
  int maxsz = 0;
  for (int b = 0; b < BATCH; b++) {
    const int* st = stats + b * 8;
    int cnt = st[0], sz;
    if (cnt == 0) { cy[b] = (float)(HH / 2); cx[b] = (float)(WWW / 2); sz = 16; }
    else {
      cy[b] = (float)st[1] / (float)cnt; cx[b] = (float)st[2] / (float)cnt;
      int hy = st[4] - st[3] + 1, hx = st[6] - st[5] + 1;
      sz = hy > hx ? hy : hx;
    }
    if (sz > maxsz) maxsz = sz;
  }
  int scl[8]; int len = 0;
  int c = maxsz < 28 ? maxsz : 28;
  while (c >= 8 && len < 8) { scl[len++] = c; c >>= 1; }
  if (len == 0 || scl[len - 1] > 8) scl[len++] = 8;
  int ns = len < NSTG ? len : NSTG;
  for (int i = 0; i < NSTG; i++) {
    int s = (i < len) ? scl[i] : 8;
    if (s > HH) s = HH;
    plan[i] = (i < ns) ? 1 : 0;
    plan[3 + i] = s;
    for (int b = 0; b < BATCH; b++) {
      int yt = (int)(cy[b] - (float)s); if (yt < 0) yt = 0;
      int yb = (int)cy[b]; if (yb > HH - s) yb = HH - s;
      int xl = (int)(cx[b] - (float)s); if (xl < 0) xl = 0;
      int xr = (int)cx[b]; if (xr > WWW - s) xr = WWW - s;
      int* st2 = plan + 6 + ((i * BATCH + b) * NQ) * 2;
      st2[0] = yt; st2[1] = xl; st2[2] = yt; st2[3] = xr;
      st2[4] = yb; st2[5] = xl; st2[6] = yb; st2[7] = xr;
    }
  }
}

// ---------------- weight conversion ----------------
__global__ void k_cvt(const float* __restrict__ src, unsigned short* __restrict__ dst, int n) {
  int i = blockIdx.x * blockDim.x + threadIdx.x;
  if (i < n) dst[i] = f2bfu(src[i]);
}
// [S][O][C][3][3] f32  ->  [S][tap][O][C] bf16
__global__ void k_cvt_conv(const float* __restrict__ src, unsigned short* __restrict__ dst, int n) {
  int idx = blockIdx.x * blockDim.x + threadIdx.x;
  if (idx >= n) return;
  int c = idx & 255;
  int o = (idx >> 8) & 255;
  int tap = (idx >> 16) % 9;
  int i = idx / (9 << 16);
  int dy = tap / 3, dx = tap % 3;
  size_t si = ((((size_t)(i * CH + o)) * CH + c) * 3 + dy) * 3 + dx;
  dst[idx] = f2bfu(src[si]);
}

// ---------------- generic WMMA GEMM: Y = act(Wbf[M,K] * X[K,1024] + bias) ----------------
// OUTMODE: 0 = f32 [M,N]; 1 = bf16 [M,N]; 2 = bf16 transposed [N, M] (for qT)
template <int OUTMODE, bool RELUBN>
__global__ __launch_bounds__(128) void k_gemm(const unsigned short* __restrict__ Wbf, int K,
                                              const float* __restrict__ bias,
                                              const float* __restrict__ X, int xstride,
                                              void* __restrict__ Yv, int ystride, int ldout,
                                              const int* __restrict__ act) {
  if (act && *act == 0) return;
  __shared__ unsigned short Xl[64 * 40];
  const int n0 = blockIdx.x * 64, m0 = blockIdx.y * 64, z = blockIdx.z;
  const int tid = threadIdx.x, lane = tid & 31, wave = tid >> 5;
  const int lrow = lane & 15, lgrp = lane >> 4;
  const float* Xz = X + (size_t)z * xstride;
  const int arow = m0 + wave * 16 + lrow;
  v8f acc[4];
#pragma unroll
  for (int t = 0; t < 4; t++) acc[t] = vzero();

  for (int kc = 0; kc < K; kc += 32) {
    { // stage X tile (32k x 64n) into LDS as [n][k], converting f32->bf16
      int k = tid & 31, g = tid >> 5;
      const float* src = Xz + (size_t)(kc + k) * HW + n0 + g * 16;
      if (kc + 32 < K) __builtin_prefetch(src + 32 * HW, 0, 1);
#pragma unroll
      for (int j = 0; j < 16; j++) Xl[(g * 16 + j) * 40 + k] = f2bfu(src[j]);
    }
    __syncthreads();
    v16bf af = load_afrag(Wbf + (size_t)arow * K + kc, lgrp);
#pragma unroll
    for (int t = 0; t < 4; t++) {
      v16bf bf = load_bfrag(&Xl[(t * 16 + lrow) * 40], lgrp);
      acc[t] = wmma_bf16(af, bf, acc[t]);
    }
    __syncthreads();
  }
#pragma unroll
  for (int t = 0; t < 4; t++) {
    int n = n0 + t * 16 + lrow;
#pragma unroll
    for (int r = 0; r < 8; r++) {
      int o = m0 + wave * 16 + r + 8 * lgrp;
      float v = acc[t][r] + (bias ? bias[o] : 0.f);
      if (RELUBN) v = fmaxf(v * BNM, 0.f);
      if (OUTMODE == 0)       ((float*)Yv)[(size_t)z * ystride + (size_t)o * ldout + n] = v;
      else if (OUTMODE == 1)  ((unsigned short*)Yv)[(size_t)z * ystride + (size_t)o * ldout + n] = f2bfu(v);
      else                    ((unsigned short*)Yv)[(size_t)z * ystride + (size_t)n * ldout + o] = f2bfu(v);
    }
  }
}

// ---------------- 3x3 conv as 9-tap implicit WMMA GEMM ----------------
// MODE 0: Y = relu((conv+bias)*BNM) ; MODE 1: Y = gamma*relu((conv+bias)*BNM) + beta*aux
template <int MODE>
__global__ __launch_bounds__(128) void k_conv3(const unsigned short* __restrict__ Wbf, // [9][256][256] bf16
                                               const float* __restrict__ bias,
                                               const float* __restrict__ X,
                                               const float* __restrict__ aux,
                                               const float* __restrict__ gamma,
                                               const float* __restrict__ beta, int sidx,
                                               float* __restrict__ Y,
                                               const int* __restrict__ act) {
  if (act && *act == 0) return;
  __shared__ unsigned short Xl[3 * 34 * 40]; // [dy][x+halo][c] bf16
  const int y = blockIdx.x, m0 = blockIdx.y * 64, b = blockIdx.z;
  const int tid = threadIdx.x, lane = tid & 31, wave = tid >> 5;
  const int lrow = lane & 15, lgrp = lane >> 4;
  const int arow = m0 + wave * 16 + lrow;
  v8f acc[2] = {vzero(), vzero()};

  for (int kc = 0; kc < CH; kc += 32) {
    for (int idx = tid; idx < 3 * 34 * 32; idx += 128) {
      int c = idx & 31, rem = idx >> 5, xh = rem % 34, dy = rem / 34;
      int yy = y + dy - 1, xx = xh - 1;
      float v = 0.f;
      if ((unsigned)yy < HH && (unsigned)xx < WWW)
        v = X[((size_t)(b * CH + kc + c) * HH + yy) * WWW + xx];
      Xl[(dy * 34 + xh) * 40 + c] = f2bfu(v);
    }
    __syncthreads();
#pragma unroll
    for (int dy = 0; dy < 3; dy++) {
#pragma unroll
      for (int dx = 0; dx < 3; dx++) {
        const int tap = dy * 3 + dx;
        const unsigned short* wr = Wbf + ((size_t)tap * CH + arow) * CH + kc;
        __builtin_prefetch(wr + CH, 0, 1);
        v16bf af = load_afrag(wr, lgrp);
#pragma unroll
        for (int t = 0; t < 2; t++) {
          v16bf bf = load_bfrag(&Xl[(dy * 34 + t * 16 + lrow + dx) * 40], lgrp);
          acc[t] = wmma_bf16(af, bf, acc[t]);
        }
      }
    }
    __syncthreads();
  }
  float g = 0.f, bt = 0.f;
  if (MODE == 1) { g = gamma[sidx]; bt = beta[sidx]; }
#pragma unroll
  for (int t = 0; t < 2; t++) {
    int x = t * 16 + lrow;
#pragma unroll
    for (int r = 0; r < 8; r++) {
      int o = m0 + wave * 16 + r + 8 * lgrp;
      size_t di = ((size_t)(b * CH + o) * HH + y) * WWW + x;
      float v = fmaxf((acc[t][r] + bias[o]) * BNM, 0.f);
      if (MODE == 1) v = g * v + bt * aux[di];
      Y[di] = v;
    }
  }
}

// ---------------- fused flash attention over the 4 quarters ----------------
// Omean[b,c,n] = (1/4) * sum_q V_q · softmax_m(Q^T K_q)  (scores/dmask collapse to all-ones:
// softmax rows sum to 1 so the mask-weighted scores are identical for every quarter)
__global__ __launch_bounds__(128) void k_attn(const unsigned short* __restrict__ qT, // [B][HW][64]
                                              const unsigned short* __restrict__ Kb, // [B*NQ][64][HW]
                                              const unsigned short* __restrict__ Vb, // [B*NQ][256][HW]
                                              float* __restrict__ Om,                // [B][256][HW]
                                              const int* __restrict__ act) {
  if (act && *act == 0) return;
  extern __shared__ char smem[];
  unsigned short* Vl = (unsigned short*)smem;                   // [256][64]  (offset 0: TDM target)
  unsigned short* Kl = (unsigned short*)(smem + 32768);         // [64][80]
  unsigned short* Pl = (unsigned short*)(smem + 32768 + 10240); // [4 waves][16][80]
  const int n0 = blockIdx.x * 16, b = blockIdx.y;
  const int tid = threadIdx.x, lane = tid & 31, wave = tid >> 5;
  const int lrow = lane & 15, lgrp = lane >> 4;
  unsigned short* Pw = Pl + wave * 16 * 80;

  const unsigned short* qrow = qT + ((size_t)b * HW + n0 + lrow) * CRQ;
  v16bf qa0 = load_afrag(qrow, lgrp);
  v16bf qa1 = load_afrag(qrow + 32, lgrp);

  float outAcc[4][8];
#pragma unroll
  for (int t = 0; t < 4; t++)
#pragma unroll
    for (int r = 0; r < 8; r++) outAcc[t][r] = 0.f;

  for (int qi = 0; qi < NQ; qi++) {
    const unsigned short* Kq = Kb + (size_t)(b * NQ + qi) * CRQ * HW;
    const unsigned short* Vq = Vb + (size_t)(b * NQ + qi) * CH * HW;
    float m_i[8], l_i[8];
    v8f oacc[4];
#pragma unroll
    for (int r = 0; r < 8; r++) { m_i[r] = -3.0e38f; l_i[r] = 0.f; }
#pragma unroll
    for (int t = 0; t < 4; t++) oacc[t] = vzero();

    for (int mc = 0; mc < HW; mc += 64) {
#if HAVE_TDM
      if (wave == 0) { // TDM: 2D tile (256 rows x 64 bf16) of V -> LDS [c][64] at lds_addr 0
        unsigned long long ga = (unsigned long long)(uintptr_t)(const void*)(Vq + mc);
        u32x4 g0; i32x8 g1; i32x4 g2; i32x4 g3; i32x8 g4;
        g0[0] = 1u;                       // count=1, user descriptor
        g0[1] = 0u;                       // lds_addr = 0 (dynamic-LDS base)
        g0[2] = (unsigned)(ga & 0xffffffffu);
        g0[3] = (unsigned)((ga >> 32) & 0x01ffffffu) | (2u << 30); // addr[56:32] | type=2
        g1[0] = (1 << 16);                // data_size = 2 bytes, no multicast/iterate/pad
        g1[1] = (int)(1024u << 16);       // tensor_dim0[15:0]
        g1[2] = (int)(256u << 16);        // tensor_dim1[15:0]
        g1[3] = (int)(64u << 16);         // tile_dim0 = 64
        g1[4] = 256;                      // tile_dim1 = 256, tile_dim2 = 0
        g1[5] = 1024;                     // tensor_dim0_stride[31:0]
        g1[6] = 0; g1[7] = 0;
        g2[0] = 0; g2[1] = 0; g2[2] = 0; g2[3] = 0; g3 = g2;
#pragma unroll
        for (int j = 0; j < 8; j++) g4[j] = 0;
        __builtin_amdgcn_tensor_load_to_lds(g0, g1, g2, g3, g4, 0);
        __builtin_amdgcn_s_wait_tensorcnt(0);
      }
#else
      for (int idx = tid; idx < CH * 64; idx += 128) {
        int m = idx & 63, c = idx >> 6;
        Vl[c * 64 + m] = Vq[(size_t)c * HW + mc + m];
      }
#endif
      // K chunk, transposed into [m][c] so B-fragment reads are contiguous
      for (int idx = tid; idx < CRQ * 64; idx += 128) {
        int m = idx & 63, c = idx >> 6;
        Kl[m * 80 + c] = Kq[(size_t)c * HW + mc + m];
      }
      __syncthreads();

      // S = Q^T K : 16(n) x 64(m) per wave (replicated across waves; PV dominates flops)
      v8f s[4];
#pragma unroll
      for (int t = 0; t < 4; t++) {
        s[t] = vzero();
        s[t] = wmma_bf16(qa0, load_bfrag(&Kl[(t * 16 + lrow) * 80], lgrp), s[t]);
        s[t] = wmma_bf16(qa1, load_bfrag(&Kl[(t * 16 + lrow) * 80 + 32], lgrp), s[t]);
      }
      // online softmax (rows n = r + 8*lgrp; reduce over 16 lanes of the half-wave)
      float scl[8];
#pragma unroll
      for (int r = 0; r < 8; r++) {
        float v = fmaxf(fmaxf(s[0][r], s[1][r]), fmaxf(s[2][r], s[3][r]));
#pragma unroll
        for (int off = 1; off < 16; off <<= 1) v = fmaxf(v, __shfl_xor(v, off, 32));
        float mn = fmaxf(m_i[r], v);
        scl[r] = __expf(m_i[r] - mn);
        m_i[r] = mn;
        l_i[r] *= scl[r];
      }
#pragma unroll
      for (int t = 0; t < 4; t++) {
#pragma unroll
        for (int r = 0; r < 8; r++) {
          oacc[t][r] *= scl[r];
          float p = __expf(s[t][r] - m_i[r]);
          l_i[r] += p;
          Pw[(r + 8 * lgrp) * 80 + t * 16 + lrow] = f2bfu(p);
        }
      }
      // O += P(16n x 64m) * V^T(64m x 256c): this wave owns c = wave*64 .. +63
#pragma unroll
      for (int k2 = 0; k2 < 2; k2++) {
        v16bf pf = load_afrag(&Pw[lrow * 80 + k2 * 32], lgrp);
#pragma unroll
        for (int t = 0; t < 4; t++) {
          int c = wave * 64 + t * 16 + lrow;
          v16bf vf = load_bfrag(&Vl[c * 64 + k2 * 32], lgrp);
          oacc[t] = wmma_bf16(pf, vf, oacc[t]);
        }
      }
      __syncthreads();
    }
    // normalize rows and fold the quarter mean (dmask == all-ones, see header comment)
#pragma unroll
    for (int r = 0; r < 8; r++) {
      float ls = l_i[r];
#pragma unroll
      for (int off = 1; off < 16; off <<= 1) ls += __shfl_xor(ls, off, 32);
      float inv = 0.25f / ls;
#pragma unroll
      for (int t = 0; t < 4; t++) outAcc[t][r] += oacc[t][r] * inv;
    }
  }
#pragma unroll
  for (int t = 0; t < 4; t++) {
    int c = wave * 64 + t * 16 + lrow;
#pragma unroll
    for (int r = 0; r < 8; r++) {
      int n = n0 + r + 8 * lgrp;
      Om[((size_t)b * CH + c) * HW + n] = outAcc[t][r];
    }
  }
}

// ---------------- quarter extraction + bilinear resize to 32x32 ----------------
__global__ void k_extract(const float* __restrict__ cur, float* __restrict__ quarters,
                          const int* __restrict__ plan, int stage) {
  if (plan[stage] == 0) return;
  const int s = plan[3 + stage];
  int bqc = blockIdx.x;
  int c = bqc % CH; int bq = bqc / CH; int qi = bq % NQ; int b = bq / NQ;
  const int* st = plan + 6 + ((stage * BATCH + b) * NQ + qi) * 2;
  const int yt = st[0], xl = st[1];
  const float* src = cur + ((size_t)b * CH + c) * HW;
  float* dst = quarters + ((size_t)bq * CH + c) * HW;
  const float sc = (float)s * (1.0f / 32.0f);
  for (int p = threadIdx.x; p < HW; p += blockDim.x) {
    int y = p >> 5, x = p & 31;
    float fy = (y + 0.5f) * sc - 0.5f;
    float fx = (x + 0.5f) * sc - 0.5f;
    int y0 = (int)floorf(fy), x0 = (int)floorf(fx);
    float wy = fy - (float)y0, wx = fx - (float)x0;
    int y0c = min(max(y0, 0), s - 1), y1c = min(max(y0 + 1, 0), s - 1);
    int x0c = min(max(x0, 0), s - 1), x1c = min(max(x0 + 1, 0), s - 1);
    const float* r0 = src + (yt + y0c) * WWW + xl;
    const float* r1 = src + (yt + y1c) * WWW + xl;
    float v = (1.f - wy) * ((1.f - wx) * r0[x0c] + wx * r0[x1c]) +
              wy * ((1.f - wx) * r1[x0c] + wx * r1[x1c]);
    dst[p] = v;
  }
}

// ---------------- launch ----------------
extern "C" void kernel_launch(void* const* d_in, const int* in_sizes, int n_in,
                              void* d_out, int out_size, void* d_ws, size_t ws_size,
                              hipStream_t stream) {
  (void)in_sizes; (void)n_in; (void)out_size; (void)ws_size;
  const float* features = (const float*)d_in[0];
  const float* mask     = (const float*)d_in[1];
  const float* adapt_w  = (const float*)d_in[2];
  const float* adapt_b  = (const float*)d_in[3];
  const float* qw       = (const float*)d_in[4];
  const float* qb       = (const float*)d_in[5];
  const float* kw       = (const float*)d_in[6];
  const float* kb       = (const float*)d_in[7];
  const float* vw       = (const float*)d_in[8];
  const float* vb       = (const float*)d_in[9];
  const float* post_w   = (const float*)d_in[10];
  const float* post_b   = (const float*)d_in[11];
  const float* gamma    = (const float*)d_in[12];
  const float* beta     = (const float*)d_in[13];
  const float* proc_w   = (const float*)d_in[14];
  const float* proc_b   = (const float*)d_in[15];
  const float* scale_w  = (const float*)d_in[16];
  const float* scale_b  = (const float*)d_in[17];
  float* cur = (float*)d_out;   // [B][256][32][32] lives in d_out the whole time

  char* ws = (char*)d_ws;
  size_t off = 0;
  auto alloc = [&](size_t bytes) -> char* {
    char* p = ws + off;
    off = (off + bytes + 255) & ~(size_t)255;
    return p;
  };
  int* stats = (int*)alloc(BATCH * 8 * sizeof(int));
  int* plan  = (int*)alloc(256 * sizeof(int));
  unsigned short* adapt_bf = (unsigned short*)alloc((size_t)CH * CIN * 2);
  unsigned short* qw_bf    = (unsigned short*)alloc((size_t)NSTG * CRQ * CH * 2);
  unsigned short* kw_bf    = (unsigned short*)alloc((size_t)NSTG * CRQ * CH * 2);
  unsigned short* vw_bf    = (unsigned short*)alloc((size_t)NSTG * CH * CH * 2);
  unsigned short* post_bf  = (unsigned short*)alloc((size_t)NSTG * 9 * CH * CH * 2);
  unsigned short* proc_bf  = (unsigned short*)alloc((size_t)NSTG * 9 * CH * CH * 2);
  unsigned short* scale_bf = (unsigned short*)alloc((size_t)NSTG * 9 * CH * CH * 2);
  float*          quarters = (float*)alloc((size_t)BATCH * NQ * CH * HW * 4);
  unsigned short* qT       = (unsigned short*)alloc((size_t)BATCH * HW * CRQ * 2);
  unsigned short* kbuf     = (unsigned short*)alloc((size_t)BATCH * NQ * CRQ * HW * 2);
  unsigned short* vbuf     = (unsigned short*)alloc((size_t)BATCH * NQ * CH * HW * 2);
  float*          omean    = (float*)alloc((size_t)BATCH * CH * HW * 4);
  float*          fusedb   = (float*)alloc((size_t)BATCH * CH * HW * 4);
  float*          procb    = (float*)alloc((size_t)BATCH * CH * HW * 4);

  // ---- one-time (per call) weight conversion to bf16 ----
  int n;
  n = CH * CIN;        k_cvt<<<(n + 255) / 256, 256, 0, stream>>>(adapt_w, adapt_bf, n);
  n = NSTG * CRQ * CH; k_cvt<<<(n + 255) / 256, 256, 0, stream>>>(qw, qw_bf, n);
  n = NSTG * CRQ * CH; k_cvt<<<(n + 255) / 256, 256, 0, stream>>>(kw, kw_bf, n);
  n = NSTG * CH * CH;  k_cvt<<<(n + 255) / 256, 256, 0, stream>>>(vw, vw_bf, n);
  n = NSTG * 9 * CH * CH;
  k_cvt_conv<<<(n + 255) / 256, 256, 0, stream>>>(post_w, post_bf, n);
  k_cvt_conv<<<(n + 255) / 256, 256, 0, stream>>>(proc_w, proc_bf, n);
  k_cvt_conv<<<(n + 255) / 256, 256, 0, stream>>>(scale_w, scale_bf, n);

  // ---- device-side mask analysis + stage plan ----
  k_stats<<<BATCH, 256, 0, stream>>>(mask, stats);
  k_plan<<<1, 32, 0, stream>>>(stats, plan);

  // ---- adapt: cur = relu(BN * (adapt_w @ features + b)) ----
  k_gemm<0, true><<<dim3(16, 4, BATCH), 128, 0, stream>>>(
      adapt_bf, CIN, adapt_b, features, CIN * HW, cur, CH * HW, HW, nullptr);

  // ---- stages (fixed 3 launches; inactive stages guarded by device flag) ----
  for (int i = 0; i < NSTG; i++) {
    const int* act = plan + i;
    k_extract<<<BATCH * NQ * CH, 256, 0, stream>>>(cur, quarters, plan, i);
    // q projection -> qT [n][c] bf16
    k_gemm<2, false><<<dim3(16, 1, BATCH), 128, 0, stream>>>(
        qw_bf + (size_t)i * CRQ * CH, CH, qb + i * CRQ, cur, CH * HW, qT, HW * CRQ, CRQ, act);
    // k projection per (b, quarter)
    k_gemm<1, false><<<dim3(16, 1, BATCH * NQ), 128, 0, stream>>>(
        kw_bf + (size_t)i * CRQ * CH, CH, kb + i * CRQ, quarters, CH * HW, kbuf, CRQ * HW, HW, act);
    // v projection per (b, quarter)
    k_gemm<1, false><<<dim3(16, 4, BATCH * NQ), 128, 0, stream>>>(
        vw_bf + (size_t)i * CH * CH, CH, vb + i * CH, quarters, CH * HW, vbuf, CH * HW, HW, act);
    // fused quarter attention -> omean
    k_attn<<<dim3(HW / 16, BATCH), 128, 53248, stream>>>(qT, kbuf, vbuf, omean, act);
    // post conv + gamma/beta residual
    k_conv3<1><<<dim3(HH, 4, BATCH), 128, 0, stream>>>(
        post_bf + (size_t)i * 9 * CH * CH, post_b + i * CH, omean, cur, gamma, beta, i, fusedb, act);
    // proc conv
    k_conv3<0><<<dim3(HH, 4, BATCH), 128, 0, stream>>>(
        proc_bf + (size_t)i * 9 * CH * CH, proc_b + i * CH, fusedb, nullptr, nullptr, nullptr, 0, procb, act);
    // scale conv -> cur (in d_out); inactive stage leaves cur untouched
    k_conv3<0><<<dim3(HH, 4, BATCH), 128, 0, stream>>>(
        scale_bf + (size_t)i * 9 * CH * CH, scale_b + i * CH, procb, nullptr, nullptr, nullptr, 0, cur, act);
  }
}
